// fstgcn_773094114080
// MI455X (gfx1250) — matrix operand order repaired and checked by reading, not compiled, not run on hardware
//
#include <hip/hip_runtime.h>
#include <hip/hip_bf16.h>

// ---------------------------------------------------------------------------
// Problem constants
// ---------------------------------------------------------------------------
#define NB   8      // batch
#define TT   10     // data time steps
#define T0   12     // original_data time steps
#define NN_  3000   // nodes
#define CC   64     // channels
#define WW   8      // windows
#define KK_  3      // GLU iterations

// Padded GEMM dims: C(3072x512) = A(3072x3008) @ B(3008x512), bf16 in / f32 acc
#define MP   3072   // node rows padded (24 * 128)
#define KP   3008   // node cols / K padded (94 * 32)
#define NP   512    // B*C

typedef __attribute__((ext_vector_type(16))) __bf16 v16bf;
typedef __attribute__((ext_vector_type(8)))  __bf16 v8bf;
typedef __attribute__((ext_vector_type(8)))  float  v8f;

// A fragment (16x32 bf16): lane half h=(lane>>4): K = {8h..8h+7} in v0-3, {8h+16..8h+23} in v4-7
__device__ __forceinline__ v16bf load_a_frag(const __bf16* p) {
  v8bf lo = *(const v8bf*)(p);        // K  +0.. +7
  v8bf hi = *(const v8bf*)(p + 16);   // K +16..+23
  return __builtin_shufflevector(lo, hi, 0,1,2,3,4,5,6,7,8,9,10,11,12,13,14,15);
}
// B fragment (32x16 bf16): lane half h: K = {8h..8h+7} in v0-3, {8h+16..8h+23} in v4-7
__device__ __forceinline__ v16bf load_b_frag(const __bf16* p) {
  v8bf lo = *(const v8bf*)(p);
  v8bf hi = *(const v8bf*)(p + 16);
  return __builtin_shufflevector(lo, hi, 0,1,2,3,4,5,6,7,8,9,10,11,12,13,14,15);
}

// ---------------------------------------------------------------------------
// WMMA GEMM: Cout(MPxNP,f32) = [Cin +] A(MPxKP,bf16) @ B^T  (B stored [col][k])
// grid (24, 2), block 256 (8 waves): block tile 128x256, wave tile 64x64
// ---------------------------------------------------------------------------
__global__ __launch_bounds__(256) void mm_wmma(const __bf16* __restrict__ A,
                                               const __bf16* __restrict__ Bt,
                                               const float*  __restrict__ Cin,
                                               float*        __restrict__ Cout) {
  const int lane  = threadIdx.x & 31;
  const int wave  = threadIdx.x >> 5;          // 0..7
  const int wm    = wave >> 2;                 // 0..1
  const int wn    = wave & 3;                  // 0..3
  const int rowBase = blockIdx.x * 128 + wm * 64;
  const int colBase = blockIdx.y * 256 + wn * 64;
  const int lrow  = lane & 15;
  const int half  = lane >> 4;

  v8f acc[4][4];
#pragma unroll
  for (int i = 0; i < 4; i++)
#pragma unroll
    for (int j = 0; j < 4; j++)
#pragma unroll
      for (int r = 0; r < 8; r++) acc[i][j][r] = 0.f;

  const __bf16* aBase = A  + (size_t)(rowBase + lrow) * KP + half * 8;
  const __bf16* bBase = Bt + (size_t)(colBase + lrow) * KP + half * 8;

  for (int kk = 0; kk < KP; kk += 32) {
    v16bf af[4], bf[4];
#pragma unroll
    for (int i = 0; i < 4; i++) af[i] = load_a_frag(aBase + (size_t)(i * 16) * KP + kk);
#pragma unroll
    for (int j = 0; j < 4; j++) bf[j] = load_b_frag(bBase + (size_t)(j * 16) * KP + kk);
#pragma unroll
    for (int i = 0; i < 4; i++)
#pragma unroll
      for (int j = 0; j < 4; j++)
        acc[i][j] = __builtin_amdgcn_wmma_f32_16x16x32_bf16(
            false, af[i], false, bf[j], (short)0, acc[i][j], false, false);
  }

  // Epilogue: VGPR r of a tile holds (M = crow0 + i*16 + r, N = colBase + j*16 + lrow).
  // One base address per i-group; all 32 stores in the group use immediate offsets
  // (r*NP floats = r*2048B row step, j*16 floats = j*64B col step).
  const int crow0 = rowBase + half * 8;
  if (Cin != nullptr) {
#pragma unroll
    for (int i = 0; i < 4; i++) {
      size_t base = (size_t)(crow0 + i * 16) * NP + (colBase + lrow);
      float* __restrict__ cp = Cout + base;
      const float* __restrict__ ip = Cin + base;
#pragma unroll
      for (int j = 0; j < 4; j++)
#pragma unroll
        for (int r = 0; r < 8; r++)
          cp[r * NP + j * 16] = acc[i][j][r] + ip[r * NP + j * 16];
    }
  } else {
#pragma unroll
    for (int i = 0; i < 4; i++) {
      size_t base = (size_t)(crow0 + i * 16) * NP + (colBase + lrow);
      float* __restrict__ cp = Cout + base;
#pragma unroll
      for (int j = 0; j < 4; j++)
#pragma unroll
        for (int r = 0; r < 8; r++)
          cp[r * NP + j * 16] = acc[i][j][r];
    }
  }
}

// ---------------------------------------------------------------------------
// Prep: adj_st (3000x9000 f32) -> 3 padded bf16 blocks A_s (MP x KP)
// ---------------------------------------------------------------------------
__global__ __launch_bounds__(256) void prep_adj(const float* __restrict__ adj,
                                                __bf16* __restrict__ Abf) {
  long long t = (long long)blockIdx.x * 256 + threadIdx.x;
  if (t >= (long long)3 * MP * KP) return;
  int m = (int)(t % KP);
  long long r = t / KP;
  int n = (int)(r % MP);
  int s = (int)(r / MP);
  float v = (n < NN_ && m < NN_) ? adj[(size_t)n * (3 * NN_) + s * NN_ + m] : 0.f;
  Abf[t] = (__bf16)v;
}

// ---------------------------------------------------------------------------
// Prep: XT[t][b*64+c][n] = bf16(data[b,t,n,c] + temb[t,c] + semb[n,c]); pad n -> 0
// ---------------------------------------------------------------------------
__global__ __launch_bounds__(256) void prep_x(const float* __restrict__ data,
                                              const float* __restrict__ temb,
                                              const float* __restrict__ semb,
                                              __bf16* __restrict__ XT) {
  long long t = (long long)blockIdx.x * 256 + threadIdx.x;
  if (t >= (long long)TT * NP * KP) return;
  int m = (int)(t % KP);
  long long r = t / KP;
  int bc = (int)(r % NP);
  int ts = (int)(r / NP);
  int b = bc >> 6, c = bc & 63;
  float v = 0.f;
  if (m < NN_)
    v = data[(((size_t)b * TT + ts) * NN_ + m) * CC + c] + temb[ts * CC + c]
        + semb[(size_t)m * CC + c];
  XT[t] = (__bf16)v;
}

// Zero the K-pad columns of DT so later WMMA reads are clean.
__global__ __launch_bounds__(256) void prep_dt_pad(__bf16* __restrict__ DT) {
  int t = blockIdx.x * 256 + threadIdx.x;           // 8*512*8
  if (t >= WW * NP * (KP - NN_)) return;
  int i = t & 7;
  int bc = (t >> 3) & (NP - 1);
  int w = t >> 12;
  DT[((size_t)w * NP + bc) * KP + NN_ + i] = (__bf16)0.f;
}

// ---------------------------------------------------------------------------
// Gated convs -> data_res written directly into d_out (B,W,N,C)
// ---------------------------------------------------------------------------
__global__ __launch_bounds__(256) void conv_kernel(const float* __restrict__ data,
                                                   const float* __restrict__ odata,
                                                   const float* __restrict__ w1,
                                                   const float* __restrict__ b1,
                                                   const float* __restrict__ w2,
                                                   const float* __restrict__ b2,
                                                   float* __restrict__ out) {
  int t = blockIdx.x * 256 + threadIdx.x;           // (b,w,n,c) c-fastest
  if (t >= NB * WW * NN_ * CC) return;
  int c = t & 63;
  int n = (t >> 6) % NN_;
  int rem = t / (CC * NN_);
  int w = rem & 7, b = rem >> 3;

  float z1l = b1[c], z1h = b1[c + 64];
#pragma unroll
  for (int m = 0; m < 3; m++) {
    const float* dp = data + (((size_t)b * TT + w + m) * NN_ + n) * CC;
    const float* wl = w1 + (size_t)c * 192 + m;
    const float* wh = w1 + (size_t)(c + 64) * 192 + m;
    for (int ic = 0; ic < CC; ic++) {
      float dv = dp[ic];
      z1l += dv * wl[ic * 3];
      z1h += dv * wh[ic * 3];
    }
  }
  float g1 = z1h / (1.f + __expf(-z1l));            // sigmoid(lhs) * rhs

  float z2l = b2[c], z2h = b2[c + 64];
#pragma unroll
  for (int m = 0; m < 5; m++) {
    const float* dp = odata + (((size_t)b * T0 + w + m) * NN_ + n) * CC;
    const float* wl = w2 + (size_t)c * 320 + m;
    const float* wh = w2 + (size_t)(c + 64) * 320 + m;
    for (int ic = 0; ic < CC; ic++) {
      float dv = dp[ic];
      z2l += dv * wl[ic * 5];
      z2h += dv * wh[ic * 5];
    }
  }
  float g2 = z2h / (1.f + __expf(-z2l));
  out[t] = g1 + g2;
}

// ---------------------------------------------------------------------------
// GLU: z = y @ glu_w[w,k] + glu_b[w,k]; d = lhs * sigmoid(rhs);
// running max into Dmax; repack d (transposed, bf16) for the next WMMA round.
// thread order: ((w*3000+n)*8+b)*64+c
// ---------------------------------------------------------------------------
__global__ __launch_bounds__(256) void glu_kernel(const float* __restrict__ Y,
                                                  const float* __restrict__ gw,
                                                  const float* __restrict__ gb,
                                                  float* __restrict__ Dmax,
                                                  __bf16* __restrict__ DT, int k) {
  int t = blockIdx.x * 256 + threadIdx.x;
  if (t >= WW * NN_ * NB * CC) return;
  int c = t & 63;
  int b = (t >> 6) & 7;
  int n = (t >> 9) % NN_;
  int w = t / (NN_ * NP);

  const float* y  = Y + (size_t)w * ((size_t)MP * NP) + ((size_t)n * NP + b * CC);
  const float* wp = gw + ((size_t)w * KK_ + k) * CC * (2 * CC);
  float zl = gb[((size_t)w * KK_ + k) * (2 * CC) + c];
  float zh = gb[((size_t)w * KK_ + k) * (2 * CC) + 64 + c];
  for (int cp = 0; cp < CC; cp++) {
    float yv = y[cp];
    zl += yv * wp[cp * 128 + c];
    zh += yv * wp[cp * 128 + 64 + c];
  }
  float d = zl / (1.f + __expf(-zh));               // lhs * sigmoid(rhs)

  float m = (k == 0) ? d : fmaxf(Dmax[t], d);
  Dmax[t] = m;
  if (k < 2) DT[(size_t)w * ((size_t)NP * KP) + ((size_t)(b * CC + c)) * KP + n] = (__bf16)d;
}

// ---------------------------------------------------------------------------
// Final: out[b,w,n,c] = data_res(already in out) + Dmax[w,n,b,c] + res
// res = data[b,w+2,n,c] + temb[w+2,c] + semb[n,c]   (f32 path, no bf16 rounding)
// ---------------------------------------------------------------------------
__global__ __launch_bounds__(256) void final_kernel(const float* __restrict__ data,
                                                    const float* __restrict__ temb,
                                                    const float* __restrict__ semb,
                                                    const float* __restrict__ Dmax,
                                                    float* __restrict__ out) {
  int t = blockIdx.x * 256 + threadIdx.x;           // (b,w,n,c)
  if (t >= NB * WW * NN_ * CC) return;
  int c = t & 63;
  int n = (t >> 6) % NN_;
  int rem = t / (CC * NN_);
  int w = rem & 7, b = rem >> 3;
  float res = data[(((size_t)b * TT + w + 2) * NN_ + n) * CC + c]
            + temb[(w + 2) * CC + c] + semb[(size_t)n * CC + c];
  float dm = Dmax[(((size_t)w * NN_ + n) * NB + b) * CC + c];
  out[t] = out[t] + dm + res;
}

// ---------------------------------------------------------------------------
// Host launcher
// ---------------------------------------------------------------------------
extern "C" void kernel_launch(void* const* d_in, const int* in_sizes, int n_in,
                              void* d_out, int out_size, void* d_ws, size_t ws_size,
                              hipStream_t stream) {
  const float* odata = (const float*)d_in[0];   // (8,12,3000,64)
  const float* data  = (const float*)d_in[1];   // (8,10,3000,64)
  const float* adj   = (const float*)d_in[2];   // (3000,9000)
  const float* w1    = (const float*)d_in[3];
  const float* b1    = (const float*)d_in[4];
  const float* w2    = (const float*)d_in[5];
  const float* b2    = (const float*)d_in[6];
  const float* temb  = (const float*)d_in[7];
  const float* semb  = (const float*)d_in[8];
  const float* gw    = (const float*)d_in[9];
  const float* gb    = (const float*)d_in[10];
  float* out = (float*)d_out;

  const size_t ASZ = (size_t)MP * KP;          // elems per A_s
  const size_t XSZ = (size_t)NP * KP;          // elems per XT[t] / DT[w]
  const size_t YSZ = (size_t)MP * NP;          // elems per Y[w]

  char* p = (char*)d_ws;
  __bf16* Abf  = (__bf16*)p;  p += 3 * ASZ * sizeof(__bf16);
  __bf16* XT   = (__bf16*)p;  p += (size_t)TT * XSZ * sizeof(__bf16);
  __bf16* DT   = (__bf16*)p;  p += (size_t)WW * XSZ * sizeof(__bf16);
  float*  Ybase = (float*)p;  p += (size_t)WW * YSZ * sizeof(float);
  float*  Ycur  = (float*)p;  p += (size_t)WW * YSZ * sizeof(float);
  float*  Dmax  = (float*)p;

  // --- prep ---
  prep_adj<<<(unsigned)((3LL * MP * KP + 255) / 256), 256, 0, stream>>>(adj, Abf);
  prep_x<<<(unsigned)(((long long)TT * NP * KP + 255) / 256), 256, 0, stream>>>(data, temb, semb, XT);
  prep_dt_pad<<<(WW * NP * (KP - NN_) + 255) / 256, 256, 0, stream>>>(DT);
  conv_kernel<<<(NB * WW * NN_ * CC + 255) / 256, 256, 0, stream>>>(data, odata, w1, b1, w2, b2, out);

  dim3 mmGrid(MP / 128, NP / 256);

  // --- Ybase[w] = A0 @ X_w + A1 @ X_{w+1}  (k-invariant wing contribution) ---
  for (int w = 0; w < WW; w++) {
    mm_wmma<<<mmGrid, 256, 0, stream>>>(Abf,            XT + (size_t)w * XSZ,
                                        nullptr,        Ybase + (size_t)w * YSZ);
    mm_wmma<<<mmGrid, 256, 0, stream>>>(Abf + ASZ,      XT + (size_t)(w + 1) * XSZ,
                                        Ybase + (size_t)w * YSZ, Ybase + (size_t)w * YSZ);
  }

  // --- GLU iterations: Ycur[w] = Ybase[w] + A2 @ (X_{w+2} | D) ; then GLU ---
  for (int k = 0; k < KK_; k++) {
    for (int w = 0; w < WW; w++) {
      const __bf16* Bk = (k == 0) ? (XT + (size_t)(w + 2) * XSZ)
                                  : (DT + (size_t)w * XSZ);
      mm_wmma<<<mmGrid, 256, 0, stream>>>(Abf + 2 * ASZ, Bk,
                                          Ybase + (size_t)w * YSZ,
                                          Ycur + (size_t)w * YSZ);
    }
    glu_kernel<<<(WW * NN_ * NB * CC + 255) / 256, 256, 0, stream>>>(Ycur, gw, gb, Dmax, DT, k);
  }

  final_kernel<<<(NB * WW * NN_ * CC + 255) / 256, 256, 0, stream>>>(data, temb, semb, Dmax, out);
}
